// InteractionBlockTripletsOnly_75797582840830
// MI455X (gfx1250) — matrix operand order repaired and checked by reading, not compiled, not run on hardware
//
#include <hip/hip_runtime.h>
#include <hip/hip_bf16.h>

typedef __attribute__((ext_vector_type(16))) __bf16 v16bf;
typedef __attribute__((ext_vector_type(8)))  __bf16 v8bf;
typedef __attribute__((ext_vector_type(4)))  __bf16 v4bf;
typedef __attribute__((ext_vector_type(8)))  float  v8f;
typedef __attribute__((ext_vector_type(4)))  int    v4i;

#define INV_SQRT2 0.70710678118654752f
#define SILU_SCALE (1.0f / 0.6f)

// Async global->LDS copy path (CDNA5 ASYNCcnt), guarded so the source always
// compiles even if this toolchain does not expose the builtin.
#if defined(__HIP_DEVICE_COMPILE__) && __has_builtin(__builtin_amdgcn_global_load_async_to_lds_b128)
#define HAVE_ASYNC_LDS 1
#else
#define HAVE_ASYNC_LDS 0
#endif

__device__ __forceinline__ void wait_async_lds() {
#if defined(__HIP_DEVICE_COMPILE__)
#if __has_builtin(__builtin_amdgcn_s_wait_asynccnt)
  __builtin_amdgcn_s_wait_asynccnt(0);
#elif HAVE_ASYNC_LDS
  asm volatile("s_wait_asynccnt 0" ::: "memory");
#endif
#endif
}

__device__ __forceinline__ float scaled_silu(float x) {
  return (x / (1.0f + __expf(-x))) * SILU_SCALE;
}

// ---------------------------------------------------------------------------
// Weight convert + transpose: Wt[n*K + k] = bf16(W[k*N + n])   (layout [N][K])
// ---------------------------------------------------------------------------
__global__ __launch_bounds__(256) void wcvt_t(const float* __restrict__ W,
                                              __bf16* __restrict__ Wt,
                                              int K, int N) {
  size_t i = (size_t)blockIdx.x * blockDim.x + threadIdx.x;
  if (i >= (size_t)K * N) return;
  int n = (int)(i / (size_t)K);
  int k = (int)(i % (size_t)K);
  Wt[i] = (__bf16)W[(size_t)k * N + n];
}

// ---------------------------------------------------------------------------
// WMMA GEMM: Out[M,N] = epilogue( A[M,K] @ Wt^T )   (Wt is bf16 [N][K])
//   MODE 0: Out = act?(acc)
//   MODE 1: Out = (aux + act?(acc)) * 1/sqrt2        (residual 2nd dense)
//   MODE 2: Out = act?(acc) * aux                    (mlp_rbf elementwise mul)
//   MODE 3: Out = Out + acc                          (concat accumulation)
//   MODE 4: Out = act(Out + acc)                     (concat final chunk)
// GATHER: A row m is A[gidx[m]]   ABF16: A is bf16 (rW buffer)
// Block: 256 threads = 8 waves; tile 128(M) x 64(N), K-step 32.
// ---------------------------------------------------------------------------
template <int ACT, int MODE, bool GATHER, bool ABF16>
__global__ __launch_bounds__(256) void gemm_wmma(
    const void* __restrict__ Aptr, const __bf16* __restrict__ Wt,
    float* __restrict__ Out, const float* __restrict__ aux,
    const int* __restrict__ gidx, int M, int N, int K, int lda) {
  __shared__ alignas(64) __bf16 sA[128 * 32];
  __shared__ alignas(64) __bf16 sB[64 * 32];
  const int tid  = threadIdx.x;
  const int wave = tid >> 5;
  const int lane = tid & 31;
  const int half = lane >> 4;
  const int l15  = lane & 15;
  const int blockM = blockIdx.y * 128;
  const int blockN = blockIdx.x * 64;

  v8f acc[4];
  for (int s = 0; s < 4; ++s)
    for (int j = 0; j < 8; ++j) acc[s][j] = 0.0f;

  for (int k0 = 0; k0 < K; k0 += 32) {
    __syncthreads();

    // ---- stage A tile 128x32 as bf16: 4 quads (float4/v4bf) per thread ----
    for (int it = 0; it < 4; ++it) {
      int q  = tid + it * 256;   // quad id, 1024 quads
      int r  = q >> 3;           // 8 quads per row
      int cq = (q & 7) * 4;
      int gm = blockM + r;
      v4bf o;
      o[0] = (__bf16)0.0f; o[1] = (__bf16)0.0f; o[2] = (__bf16)0.0f; o[3] = (__bf16)0.0f;
      if (gm < M && (k0 + cq) < K) {
        int row = GATHER ? gidx[gm] : gm;
        if (ABF16) {
          o = *(const v4bf*)((const __bf16*)Aptr + (size_t)row * lda + k0 + cq);
        } else {
          float4 v = *(const float4*)((const float*)Aptr + (size_t)row * lda + k0 + cq);
          o[0] = (__bf16)v.x; o[1] = (__bf16)v.y; o[2] = (__bf16)v.z; o[3] = (__bf16)v.w;
        }
      }
      *(v4bf*)&sA[r * 32 + cq] = o;
    }

    // prefetch next A K-tile (emits global_prefetch_b8)
    if (k0 + 32 < K) {
      int r  = tid >> 3;
      int cq = (tid & 7) * 4;
      int gm = blockM + r;
      if (gm < M) {
        int row = GATHER ? gidx[gm] : gm;
        const void* pa =
            ABF16 ? (const void*)((const __bf16*)Aptr + (size_t)row * lda + k0 + 32 + cq)
                  : (const void*)((const float*)Aptr + (size_t)row * lda + k0 + 32 + cq);
        __builtin_prefetch(pa, 0, 1);
      }
    }

    // ---- stage B tile 64x32 bf16: one 16B chunk per thread ----
    {
      int nn = tid >> 2;          // output col within tile
      int c8 = (tid & 3) * 8;     // K offset (octet)
      if ((k0 + c8) < K) {
#if HAVE_ASYNC_LDS
        __builtin_amdgcn_global_load_async_to_lds_b128(
            (__attribute__((address_space(1))) v4i*)(Wt + (size_t)(blockN + nn) * K + k0 + c8),
            (__attribute__((address_space(3))) v4i*)(sB + nn * 32 + c8), 0, 0);
#else
        *(v8bf*)&sB[nn * 32 + c8] =
            *(const v8bf*)(Wt + (size_t)(blockN + nn) * K + k0 + c8);
#endif
      } else {
        v8bf z;
        for (int j = 0; j < 8; ++j) z[j] = (__bf16)0.0f;
        *(v8bf*)&sB[nn * 32 + c8] = z;
      }
    }
#if HAVE_ASYNC_LDS
    wait_async_lds();
#endif
    __syncthreads();

    // A fragment (16x32 bf16): lane row = l15; half selects K-half
    int arow = wave * 16 + l15;
    v16bf afrag = *(const v16bf*)&sA[arow * 32 + half * 16];

    for (int s = 0; s < 4; ++s) {
      // B fragment (32x16 bf16): lane col = l15; half selects K-half
      int bcol = s * 16 + l15;
      v16bf bfrag = *(const v16bf*)&sB[bcol * 32 + half * 16];
      acc[s] = __builtin_amdgcn_wmma_f32_16x16x32_bf16(
          false, afrag, false, bfrag, (short)0, acc[s], false, false);
    }
  }

  // C layout: VGPR v -> M = v + half*8; N = l15 (+16*s)
  for (int s = 0; s < 4; ++s) {
    int gn = blockN + s * 16 + l15;
    for (int v = 0; v < 8; ++v) {
      int gm = blockM + wave * 16 + half * 8 + v;
      if (gm < M && gn < N) {
        float r = acc[s][v];
        size_t o = (size_t)gm * N + gn;
        if (MODE == 0)      Out[o] = ACT ? scaled_silu(r) : r;
        else if (MODE == 1) Out[o] = (aux[o] + (ACT ? scaled_silu(r) : r)) * INV_SQRT2;
        else if (MODE == 2) Out[o] = (ACT ? scaled_silu(r) : r) * aux[o];
        else if (MODE == 3) Out[o] = Out[o] + r;
        else                Out[o] = scaled_silu(Out[o] + r);
      }
    }
  }
}

// ---------------------------------------------------------------------------
// Triplet core: per edge n (one wave each):
//   P = rbf_w1[n] (16x7) @ sph[n] (7x16)            -> 16x16 (VALU)
//   rW = P (pad K to 32) @ m2 (16x64 gathered rows) -> 4 bf16 WMMAs
//   store rW[n][e*16 + i] (bf16) for the W_bil GEMM
// ---------------------------------------------------------------------------
__global__ __launch_bounds__(256) void triplet_wmma(
    const float* __restrict__ rbf_w1, const float* __restrict__ sph,
    const float* __restrict__ down, const int* __restrict__ id3_ba,
    __bf16* __restrict__ rW, int E) {
  const int wave = threadIdx.x >> 5;
  const int lane = threadIdx.x & 31;
  const int n = blockIdx.x * 8 + wave;
  if (n >= E) return;  // wave-uniform exit
  const int half = lane >> 4;
  const int l15  = lane & 15;

  // P row i=l15, K = half*8 + j  (A-fragment octets, K>=16 padded to zero)
  float p[8];
  for (int j = 0; j < 8; ++j) p[j] = 0.0f;
  const float* rwp = rbf_w1 + (size_t)n * 112 + (size_t)l15 * 7;  // (16,7)
  const float4* sp4 = (const float4*)(sph + (size_t)n * 112 + half * 8);
  for (int s = 0; s < 7; ++s) {
    float a = rwp[s];
    float4 u = sp4[s * 4];
    float4 w = sp4[s * 4 + 1];
    p[0] += a * u.x; p[1] += a * u.y; p[2] += a * u.z; p[3] += a * u.w;
    p[4] += a * w.x; p[5] += a * w.y; p[6] += a * w.z; p[7] += a * w.w;
  }
  v16bf afrag;
  for (int j = 0; j < 8; ++j) { afrag[j] = (__bf16)p[j]; afrag[j + 8] = (__bf16)0.0f; }

  // gather indices as 4x int4 (64B aligned)
  const int4* ib = (const int4*)(id3_ba + (size_t)n * 16);
  int4 r0 = ib[0], r1 = ib[1], r2 = ib[2], r3 = ib[3];
  int rows[16] = {r0.x, r0.y, r0.z, r0.w, r1.x, r1.y, r1.z, r1.w,
                  r2.x, r2.y, r2.z, r2.w, r3.x, r3.y, r3.z, r3.w};

  for (int s = 0; s < 4; ++s) {
    int c = s * 16 + l15;  // trip-dim column e
    // elements 0-7: K = half*8 + j; elements 8-15 hold the K>=16 zero padding.
    float t[16];
    for (int j = 0; j < 8; ++j) t[j] = down[(size_t)rows[half * 8 + j] * 64 + c];
    for (int j = 8; j < 16; ++j) t[j] = 0.0f;
    v16bf bfrag;
    for (int j = 0; j < 16; ++j) bfrag[j] = (__bf16)t[j];

    v8f cf;
    for (int j = 0; j < 8; ++j) cf[j] = 0.0f;
    cf = __builtin_amdgcn_wmma_f32_16x16x32_bf16(
        false, afrag, false, bfrag, (short)0, cf, false, false);
    // store rW[n][c*16 + (half*8 + v)] -> contiguous 8 bf16 per lane
    v8bf st;
    for (int v = 0; v < 8; ++v) st[v] = (__bf16)cf[v];
    *(v8bf*)(rW + (size_t)n * 1024 + (size_t)c * 16 + half * 8) = st;
  }
}

// ---------------------------------------------------------------------------
// Elementwise / scatter helpers
// ---------------------------------------------------------------------------
__global__ __launch_bounds__(256) void ew_addscale(const float* __restrict__ a,
                                                   const float* __restrict__ b,
                                                   float* __restrict__ o, size_t n) {
  size_t i = (size_t)blockIdx.x * blockDim.x + threadIdx.x;
  if (i < n) o[i] = (a[i] + b[i]) * INV_SQRT2;
}

// out = (skip + (xca + xac[id_swap]) * inv) * inv
__global__ __launch_bounds__(256) void ew_combine_edge(
    const float* __restrict__ skip, const float* __restrict__ xca,
    const float* __restrict__ xac, const int* __restrict__ id_swap,
    float* __restrict__ out, size_t tot) {
  size_t i = (size_t)blockIdx.x * blockDim.x + threadIdx.x;
  if (i >= tot) return;
  size_t e = i >> 9;
  int d = (int)(i & 511);
  float v = xac[(size_t)id_swap[e] * 512 + d];
  out[i] = (skip[i] + (xca[i] + v) * INV_SQRT2) * INV_SQRT2;
}

__global__ __launch_bounds__(256) void zero_f32(float* __restrict__ p, size_t n) {
  size_t i = (size_t)blockIdx.x * blockDim.x + threadIdx.x;
  if (i < n) p[i] = 0.0f;
}

// segment_sum(m * rbw, idx_t) via global_atomic_add_f32
__global__ __launch_bounds__(256) void scatter_edges_to_atoms(
    const float* __restrict__ me, const float* __restrict__ rw,
    const int* __restrict__ idx_t, float* __restrict__ acc, size_t tot) {
  size_t i = (size_t)blockIdx.x * blockDim.x + threadIdx.x;
  if (i >= tot) return;
  size_t e = i >> 9;
  int d = (int)(i & 511);
  atomicAdd(&acc[(size_t)idx_t[e] * 512 + d], me[i] * rw[i]);
}

// ---------------------------------------------------------------------------
extern "C" void kernel_launch(void* const* d_in, const int* in_sizes, int n_in,
                              void* d_out, int out_size, void* d_ws, size_t ws_size,
                              hipStream_t stream) {
  (void)n_in; (void)out_size; (void)ws_size;

  const float* h      = (const float*)d_in[0];
  const float* m      = (const float*)d_in[1];
  const float* rbf3   = (const float*)d_in[2];
  const float* rbf_w1 = (const float*)d_in[3];
  const float* sph    = (const float*)d_in[4];
  const float* rbf_h  = (const float*)d_in[5];
  const int* id3_ba   = (const int*)d_in[6];
  // d_in[7] id3_ca, d_in[8] id3_ragged_idx: implicit (e,k) enumeration
  const int* id_swap  = (const int*)d_in[9];
  const int* idx_s    = (const int*)d_in[10];
  const int* idx_t    = (const int*)d_in[11];
  const float* W_dense_ca = (const float*)d_in[12];
  const float* W_ba       = (const float*)d_in[13];
  const float* W_rbf3     = (const float*)d_in[14];
  const float* W_down     = (const float*)d_in[15];
  const float* W_bil      = (const float*)d_in[16];
  const float* W_up_ca    = (const float*)d_in[17];
  const float* W_up_ac    = (const float*)d_in[18];
  const float* W_bs       = (const float*)d_in[19];
  const float* W_as       = (const float*)d_in[20];
  const float* W_au_rbf   = (const float*)d_in[21];
  const float* W_au_d1    = (const float*)d_in[22];
  const float* W_au_res   = (const float*)d_in[23];
  const float* W_cat      = (const float*)d_in[24];
  const float* W_res_m    = (const float*)d_in[25];

  const int E  = in_sizes[1] / 512;  // nEdges
  const int NA = in_sizes[0] / 512;  // nAtoms

  // ---- bump allocator over d_ws ----
  char* base = (char*)d_ws;
  size_t woff = 0;
  auto alloc = [&](size_t bytes) -> void* {
    woff = (woff + 255) & ~(size_t)255;
    void* p = base + woff;
    woff += bytes;
    return p;
  };
  auto wcvt = [&](const float* W, int K, int N) -> __bf16* {
    __bf16* Wt = (__bf16*)alloc((size_t)K * N * sizeof(__bf16));
    int total = K * N;
    wcvt_t<<<dim3((total + 255) / 256), dim3(256), 0, stream>>>(W, Wt, K, N);
    return Wt;
  };

  // ---- weights -> bf16 [N][K] ----
  __bf16* wt_ca    = wcvt(W_dense_ca, 512, 512);
  __bf16* wt_ba    = wcvt(W_ba, 512, 512);
  __bf16* wt_rbf3  = wcvt(W_rbf3, 16, 512);
  __bf16* wt_down  = wcvt(W_down, 512, 64);
  __bf16* wt_bil   = wcvt(W_bil, 1024, 64);  // flat (e*16+i, o) row-major
  __bf16* wt_upca  = wcvt(W_up_ca, 64, 512);
  __bf16* wt_upac  = wcvt(W_up_ac, 64, 512);
  __bf16* wt_bs[2];
  for (int i = 0; i < 2; ++i) wt_bs[i] = wcvt(W_bs + (size_t)i * 262144, 512, 512);
  __bf16* wt_as[4];
  for (int i = 0; i < 4; ++i) wt_as[i] = wcvt(W_as + (size_t)i * 262144, 512, 512);
  __bf16* wt_aurbf = wcvt(W_au_rbf, 16, 512);
  __bf16* wt_aud1  = wcvt(W_au_d1, 512, 512);
  __bf16* wt_aur[6];
  for (int i = 0; i < 6; ++i) wt_aur[i] = wcvt(W_au_res + (size_t)i * 262144, 512, 512);
  __bf16* wt_cat[3];
  for (int i = 0; i < 3; ++i) wt_cat[i] = wcvt(W_cat + (size_t)i * 262144, 512, 512);
  __bf16* wt_rm[2];
  for (int i = 0; i < 2; ++i) wt_rm[i] = wcvt(W_res_m + (size_t)i * 262144, 512, 512);

  // ---- activation scratch ----
  float* bufA    = (float*)alloc((size_t)E * 512 * 4);  // skip / x / m_cur
  float* bufB    = (float*)alloc((size_t)E * 512 * 4);  // also aliases rW(bf16)
  float* bufC    = (float*)alloc((size_t)E * 512 * 4);
  float* bufD0   = (float*)alloc((size_t)E * 64 * 4);   // x_ba down
  float* bufD1   = (float*)alloc((size_t)E * 64 * 4);   // bilinear out
  float* bufAtom = (float*)alloc((size_t)NA * 512 * 4);
  float* bufAt1  = (float*)alloc((size_t)NA * 512 * 4);
  float* bufAt2  = (float*)alloc((size_t)NA * 512 * 4);

  float* h_out = (float*)d_out;
  float* m_out = h_out + (size_t)NA * 512;

  dim3 blk(256);
  dim3 gE512(512 / 64, (E + 127) / 128);
  dim3 gE64(1, (E + 127) / 128);
  dim3 gA512(512 / 64, (NA + 127) / 128);
  size_t totE = (size_t)E * 512, totA = (size_t)NA * 512;
  int ewE = (int)((totE + 255) / 256);
  int ewA = (int)((totA + 255) / 256);

  // 1. x_ca_skip = act(m @ W_dense_ca)
  gemm_wmma<1, 0, false, false><<<gE512, blk, 0, stream>>>(m, wt_ca, bufA, nullptr, nullptr, E, 512, 512, 512);
  // 2. rbfW = rbf3 @ W_rbf3 (no act)
  gemm_wmma<0, 0, false, false><<<gE512, blk, 0, stream>>>(rbf3, wt_rbf3, bufB, nullptr, nullptr, E, 512, 16, 16);
  // 3. x_ba = act(m @ W_ba) * rbfW
  gemm_wmma<1, 2, false, false><<<gE512, blk, 0, stream>>>(m, wt_ba, bufB, bufB, nullptr, E, 512, 512, 512);
  // 4. down = act(x_ba @ W_down)  (E x 64)
  gemm_wmma<1, 0, false, false><<<gE64, blk, 0, stream>>>(bufB, wt_down, bufD0, nullptr, nullptr, E, 64, 512, 512);
  // 5. triplet core -> rW bf16 (E x 1024), aliases bufB
  triplet_wmma<<<dim3((E + 7) / 8), blk, 0, stream>>>(rbf_w1, sph, bufD0, id3_ba, (__bf16*)bufB, E);
  // 6. x = rW @ W_bil_r  (E x 64)
  gemm_wmma<0, 0, false, true><<<gE64, blk, 0, stream>>>((const void*)bufB, wt_bil, bufD1, nullptr, nullptr, E, 64, 1024, 1024);
  // 7/8. x_ca / x_ac up-projections
  gemm_wmma<1, 0, false, false><<<gE512, blk, 0, stream>>>(bufD1, wt_upca, bufB, nullptr, nullptr, E, 512, 64, 64);
  gemm_wmma<1, 0, false, false><<<gE512, blk, 0, stream>>>(bufD1, wt_upac, bufC, nullptr, nullptr, E, 512, 64, 64);
  // 9. x = (x_ca_skip + (x_ca + x_ac[id_swap]) * inv) * inv
  ew_combine_edge<<<ewE, blk, 0, stream>>>(bufA, bufB, bufC, id_swap, bufA, totE);
  // 10. residual (W_bs)
  gemm_wmma<1, 0, false, false><<<gE512, blk, 0, stream>>>(bufA, wt_bs[0], bufB, nullptr, nullptr, E, 512, 512, 512);
  gemm_wmma<1, 1, false, false><<<gE512, blk, 0, stream>>>(bufB, wt_bs[1], bufA, bufA, nullptr, E, 512, 512, 512);
  // 11. m = (m + x) * inv
  ew_addscale<<<ewE, blk, 0, stream>>>(m, bufA, bufA, totE);
  // 12. residuals (W_as x2)
  for (int i = 0; i < 2; ++i) {
    gemm_wmma<1, 0, false, false><<<gE512, blk, 0, stream>>>(bufA, wt_as[2 * i], bufB, nullptr, nullptr, E, 512, 512, 512);
    gemm_wmma<1, 1, false, false><<<gE512, blk, 0, stream>>>(bufB, wt_as[2 * i + 1], bufA, bufA, nullptr, E, 512, 512, 512);
  }
  // 13. rbfWh = rbf_h @ W_au_rbf (no act)
  gemm_wmma<0, 0, false, false><<<gE512, blk, 0, stream>>>(rbf_h, wt_aurbf, bufB, nullptr, nullptr, E, 512, 16, 16);
  // 14. segment_sum(m * rbfWh, idx_t)
  zero_f32<<<ewA, blk, 0, stream>>>(bufAtom, totA);
  scatter_edges_to_atoms<<<ewE, blk, 0, stream>>>(bufA, bufB, idx_t, bufAtom, totE);
  // 15. xa = act(acc @ W_au_d1)
  gemm_wmma<1, 0, false, false><<<gA512, blk, 0, stream>>>(bufAtom, wt_aud1, bufAt1, nullptr, nullptr, NA, 512, 512, 512);
  // 16. atom residuals x3
  for (int i = 0; i < 3; ++i) {
    gemm_wmma<1, 0, false, false><<<gA512, blk, 0, stream>>>(bufAt1, wt_aur[2 * i], bufAt2, nullptr, nullptr, NA, 512, 512, 512);
    gemm_wmma<1, 1, false, false><<<gA512, blk, 0, stream>>>(bufAt2, wt_aur[2 * i + 1], bufAt1, bufAt1, nullptr, NA, 512, 512, 512);
  }
  // 17. h_out = (h + xa) * inv
  ew_addscale<<<ewA, blk, 0, stream>>>(h, bufAt1, h_out, totA);
  // 18. m2c = act(h[idx_s]@Wc1 + h[idx_t]@Wc2 + m@Wc3)
  gemm_wmma<0, 0, true, false><<<gE512, blk, 0, stream>>>(h_out, wt_cat[0], bufB, nullptr, idx_s, E, 512, 512, 512);
  gemm_wmma<0, 3, true, false><<<gE512, blk, 0, stream>>>(h_out, wt_cat[1], bufB, nullptr, idx_t, E, 512, 512, 512);
  gemm_wmma<0, 4, false, false><<<gE512, blk, 0, stream>>>(bufA, wt_cat[2], bufB, nullptr, nullptr, E, 512, 512, 512);
  // 19. residual_m
  gemm_wmma<1, 0, false, false><<<gE512, blk, 0, stream>>>(bufB, wt_rm[0], bufC, nullptr, nullptr, E, 512, 512, 512);
  gemm_wmma<1, 1, false, false><<<gE512, blk, 0, stream>>>(bufC, wt_rm[1], bufB, bufB, nullptr, E, 512, 512, 512);
  // 20. m_out = (m_cur + m2c) * inv
  ew_addscale<<<ewE, blk, 0, stream>>>(bufA, bufB, m_out, totE);
}
